// RecurrentFormulationNet_21784074126003
// MI455X (gfx1250) — matrix operand order repaired and checked by reading, not compiled, not run on hardware
//
#include <hip/hip_runtime.h>
#include <hip/hip_bf16.h>

// ---------------------------------------------------------------------------
// Recurrent GraphSAGE net for MI455X (gfx1250).
//   - mean-aggregation: edge-parallel float4 fp32 atomics (L2-resident buffers)
//   - per-layer GEMM (agg@Wl.T + bl + x@Wr.T, fused scale/bias/ReLU) on
//     v_wmma_f32_16x16x32_bf16, fp32 accumulators
//   - weights pre-packed into WMMA B-fragment order once per call
//   - 80-row M-tiles per block: B fragments held in registers across 5
//     accumulator subtiles (5x less B traffic, 10 WMMAs in flight per wave)
// ---------------------------------------------------------------------------

#define N_NODES 50000
#define N_EDGES 800000
#define HIDDEN  128
#define T_STEPS 8            // n_timesteps is a compile-time-fixed python scalar (8)
#define MSUB    5            // 16-row subtiles per block; 50000 = 625 * 80

typedef __attribute__((ext_vector_type(16))) __bf16 bf16x16;
typedef __attribute__((ext_vector_type(8)))  __bf16 bf16x8;
typedef __attribute__((ext_vector_type(8)))  float  f32x8;

static __device__ __forceinline__ f32x8 wmma_bf16(bf16x16 a, bf16x16 b, f32x8 c) {
  // D = A(16x32 bf16) * B(32x16 bf16) + C(16x16 f32)
  return __builtin_amdgcn_wmma_f32_16x16x32_bf16(
      /*neg_a=*/false, a, /*neg_b=*/false, b,
      /*c_mod=*/(short)0, c, /*reuse_a=*/false, /*reuse_b=*/false);
}

// ------------------------------ small utility kernels ----------------------

__global__ void k_zero_f32(float* __restrict__ p, int n) {
  int i = blockIdx.x * 256 + threadIdx.x;
  if (i < n) p[i] = 0.0f;
}

__global__ void k_copy_f32(const float* __restrict__ s, float* __restrict__ d, int n) {
  int i = blockIdx.x * 256 + threadIdx.x;
  if (i < n) d[i] = s[i];
}

// degree count: one thread per edge
__global__ void k_count_deg(const int* __restrict__ dst, float* __restrict__ cnt) {
  int e = blockIdx.x * 256 + threadIdx.x;
  if (e < N_EDGES) unsafeAtomicAdd(&cnt[dst[e]], 1.0f);
}

__global__ void k_inv_deg(const float* __restrict__ cnt, float* __restrict__ inv) {
  int n = blockIdx.x * 256 + threadIdx.x;
  if (n < N_NODES) inv[n] = 1.0f / fmaxf(cnt[n], 1.0f);
}

// build x = concat(F_prev(4), mesh(8), zeros(20)) into N x 32 fp32 (K padded)
__global__ void k_build_x(const float* __restrict__ Fc, const float* __restrict__ mesh,
                          float* __restrict__ xin) {
  int gid = blockIdx.x * 256 + threadIdx.x;
  if (gid >= N_NODES * 32) return;
  int n = gid >> 5, c = gid & 31;
  float v = 0.0f;
  if (c < 4)       v = Fc[n * 4 + c];
  else if (c < 12) v = mesh[n * 8 + (c - 4)];
  xin[gid] = v;
}

// ------------------------- weight fragment packing -------------------------
// B-fragment order used here (contiguous k, same as A) for V_WMMA_F32_16X16X32_BF16 (32x16 B tile):
//   lane L: column n = L&15 ; half-slot i (0..15): k = (L>>4)*16 + i
// Flat frag index: ((ntile*KC + kc)*32 + lane)*16 + i
__global__ void k_pack_w128(const float* __restrict__ W /*128x128*/, __bf16* __restrict__ frag) {
  int gid = blockIdx.x * 256 + threadIdx.x;
  if (gid >= 8 * 4 * 32 * 16) return;
  int i    = gid & 15;
  int lane = (gid >> 4) & 31;
  int kc   = (gid >> 9) & 3;
  int w    = gid >> 11;
  int n = w * 16 + (lane & 15);
  int k = kc * 32 + (lane >> 4) * 16 + i;
  frag[gid] = (__bf16)W[n * 128 + k];
}

// enc1: W is 128x12, K padded to 32 (one K-chunk)
__global__ void k_pack_w_enc1(const float* __restrict__ W /*128x12*/, __bf16* __restrict__ frag) {
  int gid = blockIdx.x * 256 + threadIdx.x;
  if (gid >= 8 * 1 * 32 * 16) return;
  int i    = gid & 15;
  int lane = (gid >> 4) & 31;
  int w    = gid >> 9;
  int n = w * 16 + (lane & 15);
  int k = (lane >> 4) * 16 + i;
  frag[gid] = (k < 12) ? (__bf16)W[n * 12 + k] : (__bf16)0.0f;
}

// ------------------------------- edge scatter ------------------------------
// gather x[src] and atomically accumulate into agg[dst]; 4 floats / thread
template <int STRIDE, int CACT>
__global__ void k_scatter(const int* __restrict__ src, const int* __restrict__ dst,
                          const float* __restrict__ x, float* __restrict__ agg) {
  constexpr int TPE = CACT / 4;
  long gid = (long)blockIdx.x * 256 + threadIdx.x;
  long e = gid / TPE;
  if (e >= N_EDGES) return;
  int c = (int)(gid % TPE) * 4;
  int s = src[e], d = dst[e];
  const float4 v = *(const float4*)(x + (size_t)s * STRIDE + c);
  float* o = agg + (size_t)d * STRIDE + c;
  unsafeAtomicAdd(o + 0, v.x);
  unsafeAtomicAdd(o + 1, v.y);
  unsafeAtomicAdd(o + 2, v.z);
  unsafeAtomicAdd(o + 3, v.w);
}

// ------------------------------- WMMA GEMM ---------------------------------
// out[m][n] = relu( (agg[m]*inv[m]) @ Wl.T + bias + x[m] @ Wr.T ),  n in [0,128)
// block: 256 threads = 8 waves; 80-node M-tile per block (5 subtiles of 16),
// 16-col N-tile per wave. B fragments held in registers across subtiles.
// 50000 = 625*80 exactly -> no divergence, EXEC all-ones for WMMA.
template <int K>
__global__ __launch_bounds__(256) void k_sage_gemm(
    const float* __restrict__ agg, const float* __restrict__ xin,
    const float* __restrict__ inv,
    const __bf16* __restrict__ fragWl, const __bf16* __restrict__ fragWr,
    const float* __restrict__ bias, float* __restrict__ out) {
  constexpr int KC  = K / 32;     // K-chunks of 32
  constexpr int STR = K + 8;      // LDS row stride (bf16) to stagger banks
  __shared__ __bf16 sAa[16 * MSUB * STR];
  __shared__ __bf16 sAx[16 * MSUB * STR];

  const int tid = threadIdx.x;
  const int m0  = blockIdx.x * (16 * MSUB);

  // --- cooperative A-stage: fp32 -> bf16, mean-scale applied to agg ---
  {
    constexpr int EPT = K / 16;            // elements per thread per row-group
    const int col = (tid & 15) * EPT;
#pragma unroll
    for (int g = 0; g < MSUB; ++g) {
      const int row = g * 16 + (tid >> 4);
      const float s = inv[m0 + row];
      const float* ar = agg + (size_t)(m0 + row) * K + col;
      const float* xr = xin + (size_t)(m0 + row) * K + col;
#pragma unroll
      for (int j = 0; j < EPT; ++j) {
        sAa[row * STR + col + j] = (__bf16)(ar[j] * s);
        sAx[row * STR + col + j] = (__bf16)(xr[j]);
      }
    }
  }
  __syncthreads();

  const int w = tid >> 5, lane = tid & 31;
  const int r = lane & 15, hi = lane >> 4;

  // B fragments: one aligned 32B load per lane per fragment, kept in VGPRs
  // for all MSUB accumulator subtiles.
  const bf16x16* pBl = (const bf16x16*)fragWl + (size_t)w * KC * 32 + lane;
  const bf16x16* pBr = (const bf16x16*)fragWr + (size_t)w * KC * 32 + lane;
  bf16x16 Bl[KC], Br[KC];
#pragma unroll
  for (int kc = 0; kc < KC; ++kc) {
    Bl[kc] = pBl[kc * 32];
    Br[kc] = pBr[kc * 32];
  }

  const int n = w * 16 + r;
  const float b = bias[n];

#pragma unroll
  for (int ms = 0; ms < MSUB; ++ms) {
    f32x8 acc;
#pragma unroll
    for (int i = 0; i < 8; ++i) acc[i] = 0.0f;

    const int rrow = (ms * 16 + r) * STR;
#pragma unroll
    for (int kc = 0; kc < KC; ++kc) {
      // A as loaded here (contiguous k, not ISA table order): lane row = lane&15; hi lanes take K+16;
      // VGPR0..7 hold K=[hi*16, hi*16+16) (matches B fragment packing)
      const int off = kc * 32 + hi * 16;
      bf16x8 a0 = *(const bf16x8*)&sAa[rrow + off];
      bf16x8 a1 = *(const bf16x8*)&sAa[rrow + off + 8];
      bf16x16 Aa = __builtin_shufflevector(a0, a1, 0, 1, 2, 3, 4, 5, 6, 7,
                                           8, 9, 10, 11, 12, 13, 14, 15);
      bf16x8 x0 = *(const bf16x8*)&sAx[rrow + off];
      bf16x8 x1 = *(const bf16x8*)&sAx[rrow + off + 8];
      bf16x16 Ax = __builtin_shufflevector(x0, x1, 0, 1, 2, 3, 4, 5, 6, 7,
                                           8, 9, 10, 11, 12, 13, 14, 15);
      acc = wmma_bf16(Aa, Bl[kc], acc);
      acc = wmma_bf16(Ax, Br[kc], acc);
    }

    // epilogue: bias + relu; C/D layout: VGPR i -> M = i + 8*hi, N = lane&15
#pragma unroll
    for (int i = 0; i < 8; ++i) {
      float v = acc[i] + b;
      v = v > 0.0f ? v : 0.0f;
      out[(size_t)(m0 + ms * 16 + i + 8 * hi) * HIDDEN + n] = v;
    }
  }
}

// ------------------------------ output head -------------------------------
// x_new = x_prev + h @ lin_W.T + lin_b ; write out[:, t, :] ; update F_cur
__global__ void k_final(const float* __restrict__ h, float* __restrict__ Fc,
                        const float* __restrict__ linW, const float* __restrict__ linb,
                        float* __restrict__ out, int t) {
  int gid = blockIdx.x * 256 + threadIdx.x;
  if (gid >= N_NODES * 4) return;
  int n = gid >> 2, f = gid & 3;
  float acc = linb[f];
  const float* hr = h + (size_t)n * HIDDEN;
  const float* wr = linW + f * HIDDEN;
#pragma unroll 8
  for (int k = 0; k < HIDDEN; ++k) acc += hr[k] * wr[k];
  float v = Fc[gid] + acc;
  out[(size_t)n * (T_STEPS * 4) + t * 4 + f] = v;
  Fc[gid] = v;
}

// ------------------------------ host driver -------------------------------

extern "C" void kernel_launch(void* const* d_in, const int* in_sizes, int n_in,
                              void* d_out, int out_size, void* d_ws, size_t ws_size,
                              hipStream_t stream) {
  const float* F0    = (const float*)d_in[0];
  const float* mesh  = (const float*)d_in[1];
  const int*   ei    = (const int*)d_in[2];
  const int*   srcI  = ei;
  const int*   dstI  = ei + N_EDGES;
  const float* e1Wl  = (const float*)d_in[4];
  const float* e1bl  = (const float*)d_in[5];
  const float* e1Wr  = (const float*)d_in[6];
  const float* e2Wl  = (const float*)d_in[7];
  const float* e2bl  = (const float*)d_in[8];
  const float* e2Wr  = (const float*)d_in[9];
  const float* d1Wl  = (const float*)d_in[10];
  const float* d1bl  = (const float*)d_in[11];
  const float* d1Wr  = (const float*)d_in[12];
  const float* d2Wl  = (const float*)d_in[13];
  const float* d2bl  = (const float*)d_in[14];
  const float* d2Wr  = (const float*)d_in[15];
  const float* linW  = (const float*)d_in[16];
  const float* linb  = (const float*)d_in[17];
  float* out = (float*)d_out;

  // ---- workspace carve-out (~85 MB total) ----
  char* ws = (char*)d_ws;
  size_t off = 0;
  auto carve = [&](size_t bytes) -> char* {
    char* p = ws + off;
    off = (off + bytes + 255) & ~(size_t)255;
    return p;
  };
  float*  F_cur = (float*)carve((size_t)N_NODES * 4 * 4);
  float*  xin   = (float*)carve((size_t)N_NODES * 32 * 4);
  float*  agg   = (float*)carve((size_t)N_NODES * HIDDEN * 4);
  float*  h_a   = (float*)carve((size_t)N_NODES * HIDDEN * 4);
  float*  h_b   = (float*)carve((size_t)N_NODES * HIDDEN * 4);
  float*  cnt   = (float*)carve((size_t)N_NODES * 4);
  float*  inv   = (float*)carve((size_t)N_NODES * 4);
  __bf16* fE1l  = (__bf16*)carve(4096 * 2);
  __bf16* fE1r  = (__bf16*)carve(4096 * 2);
  __bf16* fE2l  = (__bf16*)carve(16384 * 2);
  __bf16* fE2r  = (__bf16*)carve(16384 * 2);
  __bf16* fD1l  = (__bf16*)carve(16384 * 2);
  __bf16* fD1r  = (__bf16*)carve(16384 * 2);
  __bf16* fD2l  = (__bf16*)carve(16384 * 2);
  __bf16* fD2r  = (__bf16*)carve(16384 * 2);
  (void)ws_size; (void)n_in; (void)in_sizes; (void)out_size;

  auto blk = [](long n) { return (unsigned)((n + 255) / 256); };

  // ---- one-time (per call) prep: state init, degrees, weight packing ----
  k_copy_f32<<<blk(N_NODES * 4), 256, 0, stream>>>(F0, F_cur, N_NODES * 4);
  k_zero_f32<<<blk(N_NODES), 256, 0, stream>>>(cnt, N_NODES);
  k_count_deg<<<blk(N_EDGES), 256, 0, stream>>>(dstI, cnt);
  k_inv_deg<<<blk(N_NODES), 256, 0, stream>>>(cnt, inv);
  k_pack_w_enc1<<<blk(4096), 256, 0, stream>>>(e1Wl, fE1l);
  k_pack_w_enc1<<<blk(4096), 256, 0, stream>>>(e1Wr, fE1r);
  k_pack_w128<<<blk(16384), 256, 0, stream>>>(e2Wl, fE2l);
  k_pack_w128<<<blk(16384), 256, 0, stream>>>(e2Wr, fE2r);
  k_pack_w128<<<blk(16384), 256, 0, stream>>>(d1Wl, fD1l);
  k_pack_w128<<<blk(16384), 256, 0, stream>>>(d1Wr, fD1r);
  k_pack_w128<<<blk(16384), 256, 0, stream>>>(d2Wl, fD2l);
  k_pack_w128<<<blk(16384), 256, 0, stream>>>(d2Wr, fD2r);

  const unsigned gemmGrid = N_NODES / (16 * MSUB);  // 625, exact

  for (int t = 0; t < T_STEPS; ++t) {
    // layer input x = concat(F_cur, mesh) padded to 32 cols
    k_build_x<<<blk((long)N_NODES * 32), 256, 0, stream>>>(F_cur, mesh, xin);

    // --- enc1 (K padded 12->32) ---
    k_zero_f32<<<blk((long)N_NODES * 32), 256, 0, stream>>>(agg, N_NODES * 32);
    k_scatter<32, 16><<<blk((long)N_EDGES * 4), 256, 0, stream>>>(srcI, dstI, xin, agg);
    k_sage_gemm<32><<<gemmGrid, 256, 0, stream>>>(agg, xin, inv, fE1l, fE1r, e1bl, h_a);

    // --- enc2 ---
    k_zero_f32<<<blk((long)N_NODES * HIDDEN), 256, 0, stream>>>(agg, N_NODES * HIDDEN);
    k_scatter<HIDDEN, HIDDEN><<<blk((long)N_EDGES * 32), 256, 0, stream>>>(srcI, dstI, h_a, agg);
    k_sage_gemm<HIDDEN><<<gemmGrid, 256, 0, stream>>>(agg, h_a, inv, fE2l, fE2r, e2bl, h_b);

    // --- dec1 ---
    k_zero_f32<<<blk((long)N_NODES * HIDDEN), 256, 0, stream>>>(agg, N_NODES * HIDDEN);
    k_scatter<HIDDEN, HIDDEN><<<blk((long)N_EDGES * 32), 256, 0, stream>>>(srcI, dstI, h_b, agg);
    k_sage_gemm<HIDDEN><<<gemmGrid, 256, 0, stream>>>(agg, h_b, inv, fD1l, fD1r, d1bl, h_a);

    // --- dec2 ---
    k_zero_f32<<<blk((long)N_NODES * HIDDEN), 256, 0, stream>>>(agg, N_NODES * HIDDEN);
    k_scatter<HIDDEN, HIDDEN><<<blk((long)N_EDGES * 32), 256, 0, stream>>>(srcI, dstI, h_a, agg);
    k_sage_gemm<HIDDEN><<<gemmGrid, 256, 0, stream>>>(agg, h_a, inv, fD2l, fD2r, d2bl, h_b);

    // --- output head + state update ---
    k_final<<<blk((long)N_NODES * 4), 256, 0, stream>>>(h_b, F_cur, linW, linb, out, t);
  }
}
